// VQ_30863634989511
// MI455X (gfx1250) — matrix operand order repaired and checked by reading, compiled, not run on hardware
//
#include <hip/hip_runtime.h>
#include <stdint.h>

// VQ-VAE vector quantization for MI455X (gfx1250, wave32, WMMA + TDM).
// Distances via V_WMMA_F32_16X16X32_F16 (f16 operands, f32 accumulate);
// codebook pre-converted to f16 once, then DMA'd to LDS by the Tensor Data
// Mover (with D# pad fields generating the bank-skewed LDS row stride);
// output gather + loss in exact f32.

typedef __attribute__((ext_vector_type(16))) _Float16 v16h;
typedef __attribute__((ext_vector_type(8)))  float    v8f;
typedef __attribute__((ext_vector_type(4)))  uint32_t v4u;
typedef __attribute__((ext_vector_type(8)))  uint32_t v8u;

#define K_CODES        512
#define D_DIM          64
#define TOKENS_PER_WG  128
#define HW             4096
#define EH_STRIDE      72   // halves: 144 B rows = 128B data + 16B pad
#define XH_STRIDE      72
#define XF_STRIDE      65

union Frag32 { uint4 q[2]; v16h v; };

// ---------------------------------------------------------------------------
// Prep: f32 codebook -> f16 copy + f32 norms, written once to workspace.
// ---------------------------------------------------------------------------
__global__ __launch_bounds__(256) void vq_prep_kernel(
    const float* __restrict__ codebook,  // [512,64] f32
    _Float16* __restrict__ ws_eh,        // [512,64] f16 (row-major, unpadded)
    float* __restrict__ ws_norm)         // [512]
{
  int k = blockIdx.x * 256 + threadIdx.x;
  if (k < K_CODES) {
    const float4* row = (const float4*)(codebook + k * D_DIM);
    float s = 0.f;
#pragma unroll
    for (int j4 = 0; j4 < D_DIM / 4; ++j4) {
      float4 v = row[j4];
      s += v.x * v.x + v.y * v.y + v.z * v.z + v.w * v.w;
      int c = j4 * 4;
      ws_eh[k * D_DIM + c + 0] = (_Float16)v.x;
      ws_eh[k * D_DIM + c + 1] = (_Float16)v.y;
      ws_eh[k * D_DIM + c + 2] = (_Float16)v.z;
      ws_eh[k * D_DIM + c + 3] = (_Float16)v.w;
    }
    ws_norm[k] = s;
  }
}

// ---------------------------------------------------------------------------
// Main kernel. FROM_WS=true: TDM-load prepped f16 codebook; else convert here.
// ---------------------------------------------------------------------------
template <bool FROM_WS>
__global__ __launch_bounds__(256) void vq_main_kernel(
    const float* __restrict__ x_in,      // [32,64,64,64] BCHW
    const float* __restrict__ codebook,  // [512,64] f32
    const _Float16* __restrict__ ws_eh,  // prepped f16 codebook (or null)
    const float* __restrict__ ws_norm,   // prepped norms (or null)
    float* __restrict__ q_out,           // [32,64,64,64] BCHW
    float* __restrict__ loss_out)        // scalar (pre-zeroed)
{
  __shared__ _Float16 eh[K_CODES * EH_STRIDE];      // f16 codebook, padded rows
  __shared__ float    enorm[K_CODES];               // ||e_k||^2, f32
  __shared__ _Float16 xh[TOKENS_PER_WG * XH_STRIDE];// f16 tokens (token-major)
  __shared__ float    xf[TOKENS_PER_WG * XF_STRIDE];// f32 tokens for loss
  __shared__ int      sidx[TOKENS_PER_WG];
  __shared__ float    wg_loss;

  const int tid = threadIdx.x;
  const int b   = blockIdx.x >> 5;          // 32 WGs per image (HW/128)
  const int t0  = (blockIdx.x & 31) << 7;   // hw base of this WG's 128 tokens

  // ---- stage codebook into LDS -------------------------------------------
  if constexpr (FROM_WS) {
    // Tensor Data Mover: wave 0 issues one descriptor-driven DMA of the
    // 512x64 f16 codebook. D# pad fields insert 4 dwords (16B) after every
    // 32 dwords (128B row) -> LDS rows of EH_STRIDE=72 halves.
    if (tid < 32) {
      uint32_t lds_base = (uint32_t)(uintptr_t)(&eh[0]);  // addr[31:0] = LDS off
      uint64_t ga       = (uint64_t)(uintptr_t)ws_eh;
      v4u g0 = {0u, 0u, 0u, 0u};
      g0[0] = 1u;                                   // count=1, user descriptor
      g0[1] = lds_base;                             // lds_addr
      g0[2] = (uint32_t)ga;                         // global_addr[31:0]
      g0[3] = (uint32_t)((ga >> 32) & 0x1FFFFFFu)   // global_addr[56:32]
            | (2u << 30);                           // type = 2 ("image")
      v8u g1 = {0u, 0u, 0u, 0u, 0u, 0u, 0u, 0u};
      g1[0] = (1u << 16)       // data_size = 2 bytes
            | (1u << 20)       // pad_enable
            | (4u << 22)       // pad_interval: 32 dwords
            | (3u << 25);      // pad_amount: 4 dwords
      g1[1] = (uint32_t)D_DIM << 16;                // tensor_dim0 = 64
      g1[2] = (uint32_t)K_CODES << 16;              // tensor_dim1 = 512
      g1[3] = (uint32_t)D_DIM << 16;                // tile_dim0 = 64
      g1[4] = (uint32_t)K_CODES;                    // tile_dim1 = 512
      g1[5] = (uint32_t)D_DIM;                      // tensor_dim0_stride = 64
      v4u g2 = {0u, 0u, 0u, 0u};
      v4u g3 = {0u, 0u, 0u, 0u};
      asm volatile("tensor_load_to_lds %0, %1, %2, %3"
                   :: "s"(g0), "s"(g1), "s"(g2), "s"(g3)
                   : "memory");
    }
    for (int k = tid; k < K_CODES; k += 256) enorm[k] = ws_norm[k];
    if (tid < 32) __builtin_amdgcn_s_wait_tensorcnt(0);
  } else {
    // Fallback: convert f32 codebook (L2-resident) in-kernel.
    for (int k = tid; k < K_CODES; k += 256) {
      const float4* row = (const float4*)(codebook + k * D_DIM);
      float s = 0.f;
#pragma unroll
      for (int j4 = 0; j4 < D_DIM / 4; ++j4) {
        float4 v = row[j4];
        s += v.x * v.x + v.y * v.y + v.z * v.z + v.w * v.w;
        int c = j4 * 4;
        eh[k * EH_STRIDE + c + 0] = (_Float16)v.x;
        eh[k * EH_STRIDE + c + 1] = (_Float16)v.y;
        eh[k * EH_STRIDE + c + 2] = (_Float16)v.z;
        eh[k * EH_STRIDE + c + 3] = (_Float16)v.w;
      }
      enorm[k] = s;
    }
  }

  // ---- stage token tile: BCHW (channel-strided) -> token-major LDS --------
  for (int e = tid; e < D_DIM * TOKENS_PER_WG; e += 256) {
    int c = e >> 7, j = e & 127;                      // coalesced 128-f32 rows
    float v = x_in[((size_t)b * D_DIM + c) * HW + t0 + j];
    xf[j * XF_STRIDE + c] = v;
    xh[j * XH_STRIDE + c] = (_Float16)v;
  }
  if (tid == 0) wg_loss = 0.f;
  __syncthreads();

  // ---- per-wave WMMA: scores for 16 tokens vs all 512 codes, fused argmin --
  const int lane = tid & 31;
  const int wv   = tid >> 5;
  const int tok  = wv * 16 + (lane & 15);   // this lane's token column (N)
  const int hi   = lane >> 4;               // half-wave select

  // B fragments (ISA 16-bit B layout: lane = column N; per half-wave the
  // halves are two 8-contiguous K chunks at +0/+8 and +16/+24 of the slab)
  Frag32 bf0, bf1;
  {
    const _Float16* xr = &xh[tok * XH_STRIDE];
    const int sub = hi * 8;
    bf0.q[0] = *(const uint4*)(xr + sub + 0);
    bf0.q[1] = *(const uint4*)(xr + sub + 16);
    bf1.q[0] = *(const uint4*)(xr + 32 + sub + 0);
    bf1.q[1] = *(const uint4*)(xr + 32 + sub + 16);
  }

  float bestVal = 3.4e38f;
  int   bestIdx = 0;
  for (int ct = 0; ct < K_CODES / 16; ++ct) {
    const int code0 = ct * 16;
    // A fragments (ISA 16-bit A layout: lane = row M; per half-wave the
    // halves are two 8-contiguous K chunks at +0/+8 and +16/+24 of the slab)
    const _Float16* er = &eh[(code0 + (lane & 15)) * EH_STRIDE];
    const int base = hi * 8;
    Frag32 af0, af1;
    af0.q[0] = *(const uint4*)(er + base +  0);
    af0.q[1] = *(const uint4*)(er + base + 16);
    af1.q[0] = *(const uint4*)(er + base + 32);
    af1.q[1] = *(const uint4*)(er + base + 48);

    v8f acc = {};
    acc = __builtin_amdgcn_wmma_f32_16x16x32_f16(false, af0.v, false, bf0.v,
                                                 (short)0, acc, false, false);
    acc = __builtin_amdgcn_wmma_f32_16x16x32_f16(false, af1.v, false, bf1.v,
                                                 (short)0, acc, false, false);
    // D layout: lane holds token N=lane%16, codes M = vgpr + 8*hi.
    // rank metric: ||e||^2 - 2 x.e  (||x||^2 constant per token)
#pragma unroll
    for (int vi = 0; vi < 8; ++vi) {
      int   code = code0 + vi + hi * 8;
      float m    = enorm[code] - 2.0f * acc[vi];
      if (m < bestVal) { bestVal = m; bestIdx = code; }
    }
  }
  // combine half-wave partners (same token, disjoint code halves)
  {
    float oV = __shfl_xor(bestVal, 16, 32);
    int   oI = __shfl_xor(bestIdx, 16, 32);
    if (oV < bestVal || (oV == bestVal && oI < bestIdx)) {
      bestVal = oV; bestIdx = oI;
    }
  }
  if (lane < 16) sidx[tok] = bestIdx;
  __syncthreads();

  // ---- gather exact f32 code rows, write BCHW output, fused loss ----------
  float lacc = 0.f;
  for (int e = tid; e < D_DIM * TOKENS_PER_WG; e += 256) {
    int c = e >> 7, j = e & 127;
    float q  = codebook[sidx[j] * D_DIM + c];  // hot 128 KB table (L0/L2)
    float xv = xf[j * XF_STRIDE + c];
    q_out[((size_t)b * D_DIM + c) * HW + t0 + j] = q;  // coalesced
    float d = q - xv;
    lacc += d * d;
  }
  atomicAdd(&wg_loss, lacc);           // ds_add_f32
  __syncthreads();
  if (tid == 0)                        // loss = 1.25 * mean((q-x)^2)
    atomicAdd(loss_out, wg_loss * (1.25f / 8388608.0f));
}

extern "C" void kernel_launch(void* const* d_in, const int* in_sizes, int n_in,
                              void* d_out, int out_size, void* d_ws, size_t ws_size,
                              hipStream_t stream) {
  (void)n_in;
  const float* x = (const float*)d_in[0];   // encoder_inputs f32
  const float* e = (const float*)d_in[1];   // embedding_weight f32
  float* out  = (float*)d_out;              // [quantized BCHW | loss]
  float* loss = out + (out_size - 1);

  hipMemsetAsync(loss, 0, sizeof(float), stream);   // capture-safe zero

  const int n_tokens = in_sizes[0] / D_DIM;         // 131072
  const int n_wg     = n_tokens / TOKENS_PER_WG;    // 1024

  const size_t eh_bytes = (size_t)K_CODES * D_DIM * sizeof(_Float16); // 64 KB
  const size_t need     = eh_bytes + (size_t)K_CODES * sizeof(float); // +2 KB

  if (d_ws != nullptr && ws_size >= need) {
    _Float16* ws_eh   = (_Float16*)d_ws;
    float*    ws_norm = (float*)((char*)d_ws + eh_bytes);
    vq_prep_kernel<<<dim3(2), dim3(256), 0, stream>>>(e, ws_eh, ws_norm);
    vq_main_kernel<true><<<dim3(n_wg), dim3(256), 0, stream>>>(
        x, e, ws_eh, ws_norm, out, loss);
  } else {
    vq_main_kernel<false><<<dim3(n_wg), dim3(256), 0, stream>>>(
        x, e, nullptr, nullptr, out, loss);
  }
}
